// Stacked_LSTM_54030688583821
// MI455X (gfx1250) — matrix operand order repaired and checked by reading, not compiled, not run on hardware
//
#include <hip/hip_runtime.h>
#include <hip/hip_bf16.h>

// Stacked 2-layer LSTM, B=1024 T=512 D=128 H=100.
// Kernel 1: precompute Gx[t][b][0:400] = x[b][t] @ W_ih1^T + (b_ih1+b_hh1)  (f16, in d_ws; needs 400 MiB)
//           C tiles staged through per-wave LDS so Gx is written with coalesced b128 stores.
// Kernel 2: persistent recurrence, 64 WGs x 16 batch rows, all recurrent weights LDS-resident (f16),
//           v_wmma_f32_16x16x32_f16 for all matmuls, fp32 gate math + cell state in registers.

typedef __attribute__((ext_vector_type(16))) _Float16 v16h;
typedef __attribute__((ext_vector_type(8)))  _Float16 v8h;
typedef __attribute__((ext_vector_type(8)))  float    v8f;

#define Bsz 1024
#define Tsz 512
#define Din 128
#define H   100
#define NG  400   // 4*H gate columns
#define K1  128   // padded K for layer-1 recurrent (h1)
#define K2  224   // padded K for layer-2 (h1|h2|pad)

union AFrag { v16h v; v8h h[2]; _Float16 e[16]; };

__device__ __forceinline__ v8f wmma16(v16h a, v16h b, v8f c) {
  // D = A(16x32 f16) * B(32x16 f16) + C(f32)
  return __builtin_amdgcn_wmma_f32_16x16x32_f16(false, a, false, b, (short)0, c, false, false);
}

// B fragment (32x16) from LDS stored [N][Kstride] f16 (B[k][n] = lds[n*ks + k]).
// ISA layout: lane<16 (N = lane) holds K{k0..k0+7, k0+16..k0+23} with k0=kc*32 (+8 for lane>=16).
__device__ __forceinline__ v16h load_b_frag(const _Float16* lds, int ks, int n0, int kc, int lane) {
  const int n = n0 + (lane & 15);
  const int k = kc * 32 + ((lane & 16) ? 8 : 0);
  AFrag b;
  b.h[0] = *reinterpret_cast<const v8h*>(lds + n * ks + k);       // 16B run
  b.h[1] = *reinterpret_cast<const v8h*>(lds + n * ks + k + 16);  // 16B run
  return b.v;
}

// A fragment (16x32) from LDS stored [row][Kstride] f16.
// ISA layout: lane<16 row=lane holds K{k0..k0+7, k0+16..k0+23} with k0=kc*32 (+8 for lane>=16).
__device__ __forceinline__ v16h load_a_frag_lds(const _Float16* lds, int ks, int kc, int lane) {
  const int row = lane & 15;
  const int k0  = kc * 32 + ((lane & 16) ? 8 : 0);
  AFrag a;
  a.h[0] = *reinterpret_cast<const v8h*>(lds + row * ks + k0);       // 16B run
  a.h[1] = *reinterpret_cast<const v8h*>(lds + row * ks + k0 + 16);  // 16B run
  return a.v;
}

// A fragment from a global f32 row (x), converting to f16 on the fly.
__device__ __forceinline__ v16h load_a_frag_x(const float* xrow, int kc, int lane) {
  const int k0 = kc * 32 + ((lane & 16) ? 8 : 0);
  const float4 q0 = *reinterpret_cast<const float4*>(xrow + k0);
  const float4 q1 = *reinterpret_cast<const float4*>(xrow + k0 + 4);
  const float4 q2 = *reinterpret_cast<const float4*>(xrow + k0 + 16);
  const float4 q3 = *reinterpret_cast<const float4*>(xrow + k0 + 20);
  AFrag a;
  a.e[0]=(_Float16)q0.x; a.e[1]=(_Float16)q0.y; a.e[2]=(_Float16)q0.z; a.e[3]=(_Float16)q0.w;
  a.e[4]=(_Float16)q1.x; a.e[5]=(_Float16)q1.y; a.e[6]=(_Float16)q1.z; a.e[7]=(_Float16)q1.w;
  a.e[8]=(_Float16)q2.x; a.e[9]=(_Float16)q2.y; a.e[10]=(_Float16)q2.z; a.e[11]=(_Float16)q2.w;
  a.e[12]=(_Float16)q3.x; a.e[13]=(_Float16)q3.y; a.e[14]=(_Float16)q3.z; a.e[15]=(_Float16)q3.w;
  return a.v;
}

// Scatter a C fragment (16x16 f32) into LDS gates [16][NG] f32.
// C layout: VGPR r -> M = r (+8 for lanes 16-31), N = lane&15.
__device__ __forceinline__ void store_c_gates(float* lG, v8f c, int n0, int lane) {
  const int n  = n0 + (lane & 15);
  const int mh = (lane & 16) ? 8 : 0;
#pragma unroll
  for (int r = 0; r < 8; ++r) lG[(mh + r) * NG + n] = c[r];
}

__device__ __forceinline__ float sigm(float x) { return 1.f / (1.f + __expf(-x)); }

// ---------------------------------------------------------------------------
// Kernel 1: Gx[t][b][n] = sum_k x[b][t][k] * W_ih1[n][k] + (b_ih1[n]+b_hh1[n])
// ---------------------------------------------------------------------------
__global__ __launch_bounds__(256, 1)
void lstm_gx_kernel(const float* __restrict__ x, const float* __restrict__ Wih1,
                    const float* __restrict__ bih1, const float* __restrict__ bhh1,
                    _Float16* __restrict__ Gx) {
  __shared__ __attribute__((aligned(32))) _Float16 lW[NG * Din];   // [n][k], 102.4 KB
  __shared__ __attribute__((aligned(32))) _Float16 lS[8 * 16 * 16];// per-wave C staging, 4 KB
  __shared__ float lb[NG];

  const int tid = threadIdx.x;
  for (int i = tid; i < NG * Din; i += 256) lW[i] = (_Float16)Wih1[i];  // same [n][k] layout
  for (int i = tid; i < NG; i += 256) lb[i] = bih1[i] + bhh1[i];
  __syncthreads();

  const int lane   = tid & 31;
  const int wave   = tid >> 5;
  const int wid    = ((blockIdx.x * blockDim.x) + tid) >> 5;
  const int nwaves = (gridDim.x * blockDim.x) >> 5;
  _Float16* st = lS + wave * 256;            // this wave's 16x16 f16 staging tile

  // tiles: 64 b-tiles x 512 t  (16 batch rows at a fixed t per tile)
  for (int tile = wid; tile < 64 * Tsz; tile += nwaves) {
    const int btile = tile >> 9;       // /512
    const int t     = tile & (Tsz - 1);
    const int b     = btile * 16 + (lane & 15);
    const float* xrow = x + ((size_t)b * Tsz + t) * Din;
    v16h a0 = load_a_frag_x(xrow, 0, lane);
    v16h a1 = load_a_frag_x(xrow, 1, lane);
    v16h a2 = load_a_frag_x(xrow, 2, lane);
    v16h a3 = load_a_frag_x(xrow, 3, lane);
    for (int nt = 0; nt < 25; ++nt) {
      const int n0 = nt * 16;
      v8f c = {};
      c = wmma16(a0, load_b_frag(lW, Din, n0, 0, lane), c);
      c = wmma16(a1, load_b_frag(lW, Din, n0, 1, lane), c);
      c = wmma16(a2, load_b_frag(lW, Din, n0, 2, lane), c);
      c = wmma16(a3, load_b_frag(lW, Din, n0, 3, lane), c);
      // scatter C+bias into per-wave LDS staging (conflict-free, in-order within wave)
      {
        const float bias = lb[n0 + (lane & 15)];
        const int   n    = lane & 15;
        const int   mh   = (lane & 16) ? 8 : 0;
#pragma unroll
        for (int r = 0; r < 8; ++r) st[(mh + r) * 16 + n] = (_Float16)(c[r] + bias);
      }
      // coalesced b128 store of the 16x16 f16 tile: lane -> (row = lane/2, 8 cols)
      {
        const int row = lane >> 1;
        const int off = (lane & 1) * 8;
        const v8h val = *reinterpret_cast<const v8h*>(st + row * 16 + off);
        _Float16* gp = Gx + (size_t)t * (Bsz * NG) + (size_t)(btile * 16 + row) * NG + n0 + off;
        *reinterpret_cast<v8h*>(gp) = val;
      }
    }
  }
}

// ---------------------------------------------------------------------------
// Kernel 2: persistent recurrence. 64 WGs x 256 threads (8 waves), 16 rows/WG.
// LDS: 320,064 B total (fits 320 KB WGP LDS).
// ---------------------------------------------------------------------------
__global__ __launch_bounds__(256, 1)
void lstm_rec_kernel(const float* __restrict__ Whh1, const float* __restrict__ Wih2,
                     const float* __restrict__ Whh2, const float* __restrict__ bih2,
                     const float* __restrict__ bhh2, const _Float16* __restrict__ Gx,
                     float* __restrict__ out) {
  __shared__ __attribute__((aligned(32))) _Float16 lW1[NG * K1];  // 102,400 B  (k>=100 zero)
  __shared__ __attribute__((aligned(32))) _Float16 lW2[NG * K2];  // 179,200 B  ([Wih2|Whh2|0])
  __shared__ __attribute__((aligned(32))) _Float16 lH1[16 * K1];  //   4,096 B  h1 padded
  __shared__ __attribute__((aligned(32))) _Float16 lHC[16 * K2];  //   7,168 B  [h1|h2|0]
  __shared__ float lG[16 * NG];                                   //  25,600 B  gate staging
  __shared__ float lb2[NG];                                       //   1,600 B

  const int tid = threadIdx.x;
  // --- one-time weight load + f16 convert (amortized over 512 steps) ---
  for (int i = tid; i < NG * K1; i += 256) {
    const int n = i >> 7, k = i & 127;
    lW1[i] = (k < H) ? (_Float16)Whh1[n * H + k] : (_Float16)0.f;
  }
  for (int i = tid; i < NG * K2; i += 256) {
    const int n = i / K2, k = i - n * K2;
    float v = 0.f;
    if (k < H)          v = Wih2[n * H + k];
    else if (k < 2 * H) v = Whh2[n * H + (k - H)];
    lW2[i] = (_Float16)v;
  }
  for (int i = tid; i < 16 * K1; i += 256) lH1[i] = (_Float16)0.f;
  for (int i = tid; i < 16 * K2; i += 256) lHC[i] = (_Float16)0.f;
  for (int i = tid; i < NG; i += 256) lb2[i] = bih2[i] + bhh2[i];

  const int lane = tid & 31;
  const int wave = tid >> 5;
  const int m0   = blockIdx.x * 16;  // batch row base

  // elementwise ownership: thread -> (row = tid>>4, cols j = (tid&15)+16*it)
  const int erow = tid >> 4;
  const int ejb  = tid & 15;
  float c1r[7], c2r[7];
#pragma unroll
  for (int i = 0; i < 7; ++i) { c1r[i] = 0.f; c2r[i] = 0.f; }

  __syncthreads();

  for (int t = 0; t < Tsz; ++t) {
    // ---- phase 1: layer-1 recurrent GEMM  lG = h1 @ W_hh1^T ----
    {
      v16h a0 = load_a_frag_lds(lH1, K1, 0, lane);
      v16h a1 = load_a_frag_lds(lH1, K1, 1, lane);
      v16h a2 = load_a_frag_lds(lH1, K1, 2, lane);
      v16h a3 = load_a_frag_lds(lH1, K1, 3, lane);
      for (int nt = wave; nt < 25; nt += 8) {
        const int n0 = nt * 16;
        v8f c = {};
        c = wmma16(a0, load_b_frag(lW1, K1, n0, 0, lane), c);
        c = wmma16(a1, load_b_frag(lW1, K1, n0, 1, lane), c);
        c = wmma16(a2, load_b_frag(lW1, K1, n0, 2, lane), c);
        c = wmma16(a3, load_b_frag(lW1, K1, n0, 3, lane), c);
        store_c_gates(lG, c, n0, lane);
      }
    }
    __syncthreads();

    // ---- phase 2: layer-1 elementwise (Gx + lG -> h1, c1) ----
    {
      const _Float16* gx = Gx + ((size_t)t * Bsz + m0 + erow) * NG;
#pragma unroll
      for (int it = 0; it < 7; ++it) {
        const int j = ejb + 16 * it;
        if (j < H) {
          const float gi = (float)gx[j]         + lG[erow * NG + j];
          const float gf = (float)gx[j + H]     + lG[erow * NG + j + H];
          const float gg = (float)gx[j + 2 * H] + lG[erow * NG + j + 2 * H];
          const float go = (float)gx[j + 3 * H] + lG[erow * NG + j + 3 * H];
          const float i_ = sigm(gi), f_ = sigm(gf), g_ = tanhf(gg), o_ = sigm(go);
          const float c = f_ * c1r[it] + i_ * g_;
          c1r[it] = c;
          const float h = o_ * tanhf(c);
          lH1[erow * K1 + j] = (_Float16)h;
          lHC[erow * K2 + j] = (_Float16)h;
        }
      }
      // prefetch next step's Gx slice for this row (global_prefetch_b8)
      if (t + 1 < Tsz && ejb < 7)
        __builtin_prefetch(gx + (size_t)Bsz * NG + ejb * 64, 0, 1);
    }
    __syncthreads();

    // ---- phase 3: layer-2 GEMM  lG = [h1|h2] @ [W_ih2|W_hh2]^T ----
    {
      v16h a[7];
#pragma unroll
      for (int kc = 0; kc < 7; ++kc) a[kc] = load_a_frag_lds(lHC, K2, kc, lane);
      for (int nt = wave; nt < 25; nt += 8) {
        const int n0 = nt * 16;
        v8f c = {};
#pragma unroll
        for (int kc = 0; kc < 7; ++kc)
          c = wmma16(a[kc], load_b_frag(lW2, K2, n0, kc, lane), c);
        store_c_gates(lG, c, n0, lane);
      }
    }
    __syncthreads();

    // ---- phase 4: layer-2 elementwise + output store ----
    {
#pragma unroll
      for (int it = 0; it < 7; ++it) {
        const int j = ejb + 16 * it;
        if (j < H) {
          const float gi = lb2[j]         + lG[erow * NG + j];
          const float gf = lb2[j + H]     + lG[erow * NG + j + H];
          const float gg = lb2[j + 2 * H] + lG[erow * NG + j + 2 * H];
          const float go = lb2[j + 3 * H] + lG[erow * NG + j + 3 * H];
          const float i_ = sigm(gi), f_ = sigm(gf), g_ = tanhf(gg), o_ = sigm(go);
          const float c = f_ * c2r[it] + i_ * g_;
          c2r[it] = c;
          const float h = o_ * tanhf(c);
          lHC[erow * K2 + H + j] = (_Float16)h;
          out[((size_t)(m0 + erow) * Tsz + t) * H + j] = h;
        }
      }
    }
    __syncthreads();
  }
}

extern "C" void kernel_launch(void* const* d_in, const int* in_sizes, int n_in,
                              void* d_out, int out_size, void* d_ws, size_t ws_size,
                              hipStream_t stream) {
  const float* x    = (const float*)d_in[0];
  const float* Wih1 = (const float*)d_in[1];
  const float* Whh1 = (const float*)d_in[2];
  const float* bih1 = (const float*)d_in[3];
  const float* bhh1 = (const float*)d_in[4];
  const float* Wih2 = (const float*)d_in[5];
  const float* Whh2 = (const float*)d_in[6];
  const float* bih2 = (const float*)d_in[7];
  const float* bhh2 = (const float*)d_in[8];
  float*     out = (float*)d_out;
  _Float16*  Gx  = (_Float16*)d_ws;  // requires T*B*400*2 = 400 MiB of workspace

  // Phase A: input-projection GEMM for all timesteps (no recurrence).
  lstm_gx_kernel<<<1024, 256, 0, stream>>>(x, Wih1, bih1, bhh1, Gx);
  // Phase B: persistent recurrent kernel, no inter-WG sync needed (state is row-local).
  lstm_rec_kernel<<<64, 256, 0, stream>>>(Whh1, Wih2, Whh2, bih2, bhh2, Gx, out);
}
